// ClusterPosteriorDistribution_75393855914516
// MI455X (gfx1250) — matrix operand order repaired and checked by reading, not compiled, not run on hardware
//
#include <hip/hip_runtime.h>
#include <hip/hip_bf16.h>

typedef __attribute__((ext_vector_type(16))) __bf16 v16bf;
typedef __attribute__((ext_vector_type(8)))  float  v8f;

#define B_ROWS 8192
#define C_ROWS 4096
#define D_DIM  1024

#define TBM 128
#define TBN 128
#define TBK 32
#define NKSTEPS (D_DIM / TBK)
#define LDT 40   // padded bf16 row stride (80 bytes -> 16B-aligned chunks)

__device__ __forceinline__ unsigned short f2bf_rne(float f) {
    unsigned int u = __float_as_uint(f);
    unsigned int r = u + 0x7FFFu + ((u >> 16) & 1u);
    return (unsigned short)(r >> 16);
}
__device__ __forceinline__ float bf2f(unsigned short h) {
    return __uint_as_float(((unsigned int)h) << 16);
}

// ---------------- Pass 0: e_sq[c] = sum_d E[c][d]^2 ----------------
__global__ __launch_bounds__(256) void esq_kernel(const float* __restrict__ E,
                                                  float* __restrict__ e_sq) {
    const int wave = threadIdx.x >> 5;
    const int lane = threadIdx.x & 31;
    const int row  = blockIdx.x * 8 + wave;
    const float4* rp = (const float4*)(E + (size_t)row * D_DIM);
    float acc = 0.f;
    #pragma unroll
    for (int i = 0; i < D_DIM / 4; i += 32) {
        float4 v = rp[i + lane];
        acc += v.x * v.x + v.y * v.y + v.z * v.z + v.w * v.w;
    }
    #pragma unroll
    for (int off = 16; off > 0; off >>= 1) acc += __shfl_xor(acc, off, 32);
    if (lane == 0) e_sq[row] = acc;
}

// ---------------- Pass 1: T[b][c] = 2 * (X[b].E[c]) - e_sq[c] ----------------
union Frag  { uint4 u[2]; v16bf v; };
union Pack8 { unsigned short s[8]; uint4 u; };

__global__ __launch_bounds__(256) void gemm_kernel(const float* __restrict__ X,
                                                   const float* __restrict__ E,
                                                   const float* __restrict__ e_sq,
                                                   float* __restrict__ T) {
    // double-buffered bf16 hi/lo tiles
    __shared__ unsigned short sAhi[2][TBM * LDT];
    __shared__ unsigned short sAlo[2][TBM * LDT];
    __shared__ unsigned short sBhi[2][TBN * LDT];
    __shared__ unsigned short sBlo[2][TBN * LDT];

    const int tid  = threadIdx.x;
    const int lane = tid & 31;
    const int wave = tid >> 5;
    const int m0 = blockIdx.y * TBM;
    const int n0 = blockIdx.x * TBN;
    const int wm = (wave & 1) * 64;    // 2 waves across M
    const int wn = (wave >> 1) * 32;   // 4 waves across N

    // staging-load mapping (both A and B tiles are 128x32 fp32, 16 elems/thread)
    const int tm = tid >> 1;           // 0..127
    const int tk = (tid & 1) * 16;     // 0 or 16

    const float* ga = X + (size_t)(m0 + tm) * D_DIM + tk;
    const float* gb = E + (size_t)(n0 + tm) * D_DIM + tk;

    v8f acc[4][2];
    #pragma unroll
    for (int i = 0; i < 4; i++)
        #pragma unroll
        for (int j = 0; j < 2; j++)
            #pragma unroll
            for (int k = 0; k < 8; k++) acc[i][j][k] = 0.f;

    float fa[16], fb[16];

    auto load_regs = [&](int k0) {
        const float* ap = ga + k0;
        const float* bp = gb + k0;
        #pragma unroll
        for (int c = 0; c < 4; c++) {
            float4 v = ((const float4*)ap)[c];
            fa[c * 4 + 0] = v.x; fa[c * 4 + 1] = v.y;
            fa[c * 4 + 2] = v.z; fa[c * 4 + 3] = v.w;
            float4 w = ((const float4*)bp)[c];
            fb[c * 4 + 0] = w.x; fb[c * 4 + 1] = w.y;
            fb[c * 4 + 2] = w.z; fb[c * 4 + 3] = w.w;
        }
        if (k0 + TBK < D_DIM) {   // WGP-scope prefetch of the tile after this one
            __builtin_prefetch(ap + TBK, 0, 3);
            __builtin_prefetch(bp + TBK, 0, 3);
        }
    };

    auto convert_store = [&](int buf) {
        #pragma unroll
        for (int half = 0; half < 2; half++) {
            Pack8 ah, al, bh, bl;
            #pragma unroll
            for (int j = 0; j < 8; j++) {
                float f = fa[half * 8 + j];
                unsigned short h = f2bf_rne(f);
                ah.s[j] = h;
                al.s[j] = f2bf_rne(f - bf2f(h));
                float g = fb[half * 8 + j];
                unsigned short gh = f2bf_rne(g);
                bh.s[j] = gh;
                bl.s[j] = f2bf_rne(g - bf2f(gh));
            }
            ((uint4*)&sAhi[buf][tm * LDT + tk])[half] = ah.u;
            ((uint4*)&sAlo[buf][tm * LDT + tk])[half] = al.u;
            ((uint4*)&sBhi[buf][tm * LDT + tk])[half] = bh.u;
            ((uint4*)&sBlo[buf][tm * LDT + tk])[half] = bl.u;
        }
    };

    // prologue: stage tile 0
    load_regs(0);
    convert_store(0);
    __syncthreads();

    for (int it = 0; it < NKSTEPS; ++it) {
        const int cur = it & 1;
        const bool has_next = (it + 1 < NKSTEPS);
        if (has_next) load_regs((it + 1) * TBK);

        // ---- fragment loads per ISA 16-bit WMMA layouts ----
        // A 16x32: lanes 0-15 -> elems[0..7]=K0..7,[8..15]=K16..23 ; lanes 16-31 -> K8..15 / K24..31
        const int arow_l = lane & 15;
        const int achunk = (lane < 16) ? 0 : 1;   // uint4 index 0/1, then +2
        Frag aH[4], aL[4], bH[2], bL[2];
        #pragma unroll
        for (int mi = 0; mi < 4; mi++) {
            const int r = wm + mi * 16 + arow_l;
            const uint4* ph = (const uint4*)&sAhi[cur][r * LDT];
            const uint4* pl = (const uint4*)&sAlo[cur][r * LDT];
            aH[mi].u[0] = ph[achunk];  aH[mi].u[1] = ph[achunk + 2];
            aL[mi].u[0] = pl[achunk];  aL[mi].u[1] = pl[achunk + 2];
        }
        // B 32x16: lanes 0-15 -> col=lane, K0..7,K16..23 ; lanes 16-31 -> col=lane-16, K8..15,K24..31
        const int ksel = (lane < 16) ? 0 : 8;
        #pragma unroll
        for (int ni = 0; ni < 2; ni++) {
            const int ncol = wn + ni * 16 + (lane & 15);
            const uint4* ph = (const uint4*)&sBhi[cur][ncol * LDT + ksel];
            const uint4* pl = (const uint4*)&sBlo[cur][ncol * LDT + ksel];
            bH[ni].u[0] = ph[0]; bH[ni].u[1] = ph[2];
            bL[ni].u[0] = pl[0]; bL[ni].u[1] = pl[2];
        }

        // ---- 4x2 tiles, 3-term bf16 split (hi*hi + hi*lo + lo*hi) = 24 WMMAs ----
        #pragma unroll
        for (int mi = 0; mi < 4; mi++)
            #pragma unroll
            for (int ni = 0; ni < 2; ni++) {
                acc[mi][ni] = __builtin_amdgcn_wmma_f32_16x16x32_bf16(
                    false, aH[mi].v, false, bH[ni].v, (short)0, acc[mi][ni], false, false);
                acc[mi][ni] = __builtin_amdgcn_wmma_f32_16x16x32_bf16(
                    false, aH[mi].v, false, bL[ni].v, (short)0, acc[mi][ni], false, false);
                acc[mi][ni] = __builtin_amdgcn_wmma_f32_16x16x32_bf16(
                    false, aL[mi].v, false, bH[ni].v, (short)0, acc[mi][ni], false, false);
            }

        // stage next tile into the other buffer while WMMAs drain
        if (has_next) convert_store(cur ^ 1);
        __syncthreads();
    }

    // ---- epilogue: T = 2*cross - e_sq (shift/scale-invariant logits form) ----
    #pragma unroll
    for (int ni = 0; ni < 2; ni++) {
        const int n  = n0 + wn + ni * 16 + (lane & 15);
        const float es = e_sq[n];
        #pragma unroll
        for (int mi = 0; mi < 4; mi++) {
            const int mbase = m0 + wm + mi * 16 + ((lane < 16) ? 0 : 8);
            #pragma unroll
            for (int r = 0; r < 8; r++) {
                T[(size_t)(mbase + r) * C_ROWS + n] = 2.f * acc[mi][ni][r] - es;
            }
        }
    }
}

// ---------------- Pass 2: per-row normalize + softmax + argmax ----------------
__global__ __launch_bounds__(256) void softmax_kernel(float* __restrict__ T,
                                                      float* __restrict__ argmax_out) {
    const int row = blockIdx.x;
    float* rp = T + (size_t)row * C_ROWS;
    const int tid  = threadIdx.x;
    const int lane = tid & 31;
    const int wave = tid >> 5;

    float vals[16];
    float s = 0.f, ss = 0.f, mx = -INFINITY;
    int mi = 0x7fffffff;
    #pragma unroll
    for (int c = 0; c < 4; c++) {
        const int vidx = tid + c * 256;        // float4 index, coalesced
        float4 v = ((const float4*)rp)[vidx];
        float tv[4] = {v.x, v.y, v.z, v.w};
        #pragma unroll
        for (int j = 0; j < 4; j++) {
            float t = tv[j];
            vals[c * 4 + j] = t;
            s += t; ss += t * t;
            int idx = vidx * 4 + j;
            if (t > mx || (t == mx && idx < mi)) { mx = t; mi = idx; }
        }
    }
    #pragma unroll
    for (int off = 16; off > 0; off >>= 1) {
        s  += __shfl_xor(s, off, 32);
        ss += __shfl_xor(ss, off, 32);
        float om = __shfl_xor(mx, off, 32);
        int   oi = __shfl_xor(mi, off, 32);
        if (om > mx || (om == mx && oi < mi)) { mx = om; mi = oi; }
    }

    __shared__ float ls[8], lss[8], lmx[8];
    __shared__ int   lmi[8];
    if (lane == 0) { ls[wave] = s; lss[wave] = ss; lmx[wave] = mx; lmi[wave] = mi; }
    __syncthreads();
    float S = 0.f, SS = 0.f, MX = -INFINITY;
    int MI = 0x7fffffff;
    #pragma unroll
    for (int w = 0; w < 8; w++) {
        S += ls[w]; SS += lss[w];
        if (lmx[w] > MX || (lmx[w] == MX && lmi[w] < MI)) { MX = lmx[w]; MI = lmi[w]; }
    }

    const float mean = S * (1.f / (float)C_ROWS);
    const float var  = fmaxf(SS * (1.f / (float)C_ROWS) - mean * mean, 1e-30f);
    const float rstd = rsqrtf(var);
    const float lmax = (MX - mean) * rstd;

    float p[16];
    float es = 0.f;
    #pragma unroll
    for (int i = 0; i < 16; i++) {
        p[i] = __expf((vals[i] - mean) * rstd - lmax);
        es += p[i];
    }
    __syncthreads();  // done reading ls[]
    #pragma unroll
    for (int off = 16; off > 0; off >>= 1) es += __shfl_xor(es, off, 32);
    if (lane == 0) ls[wave] = es;
    __syncthreads();
    float ES = 0.f;
    #pragma unroll
    for (int w = 0; w < 8; w++) ES += ls[w];
    const float inv = 1.f / ES;

    #pragma unroll
    for (int c = 0; c < 4; c++) {
        const int vidx = tid + c * 256;
        float4 o;
        o.x = p[c * 4 + 0] * inv; o.y = p[c * 4 + 1] * inv;
        o.z = p[c * 4 + 2] * inv; o.w = p[c * 4 + 3] * inv;
        ((float4*)rp)[vidx] = o;
    }
    if (tid == 0) argmax_out[row] = (float)MI;
}

extern "C" void kernel_launch(void* const* d_in, const int* in_sizes, int n_in,
                              void* d_out, int out_size, void* d_ws, size_t ws_size,
                              hipStream_t stream) {
    const float* X = (const float*)d_in[0];   // [8192, 1024]
    const float* E = (const float*)d_in[1];   // [4096, 1024]
    float* out  = (float*)d_out;              // [8192*4096] dist + [8192] argmax
    float* e_sq = (float*)d_ws;               // [4096]

    esq_kernel<<<C_ROWS / 8, 256, 0, stream>>>(E, e_sq);

    dim3 grid(C_ROWS / TBN, B_ROWS / TBM);    // (32, 64)
    gemm_kernel<<<grid, 256, 0, stream>>>(X, E, e_sq, out);

    softmax_kernel<<<B_ROWS, 256, 0, stream>>>(out, out + (size_t)B_ROWS * C_ROWS);
}
